// CNet_57346403336673
// MI455X (gfx1250) — compile-verified, matrix operand order checked by reading, not run on hardware
//
#include <hip/hip_runtime.h>
#include <hip/hip_bf16.h>

#define IN_CHN 128
#define SPAT   (48*48*48)        // 110592
#define NBATCH 2
#define NVOX   (NBATCH*SPAT)     // 221184
#define NPROJ  192
#define EPSV   1e-5f

#if defined(__has_builtin)
#if __has_builtin(__builtin_amdgcn_global_load_async_to_lds_b128)
#define USE_ASYNC_LDS 1
#endif
#endif
#ifndef USE_ASYNC_LDS
#define USE_ASYNC_LDS 0
#endif

typedef __attribute__((ext_vector_type(16))) __bf16 v16bf;
typedef __attribute__((ext_vector_type(8)))  float  v8f;
typedef int v4i_ __attribute__((vector_size(16)));   // matches async-LDS builtin param

static __device__ __forceinline__ int laneid() { return (int)(threadIdx.x & 31u); }

static __device__ __forceinline__ v8f wmma_bf16(v16bf a, v16bf b, v8f c) {
  // v_wmma_f32_16x16x32_bf16 : D = A(16x32) * B(32x16) + C
  return __builtin_amdgcn_wmma_f32_16x16x32_bf16(false, a, false, b, (short)0, c, false, false);
}

// A (or B where B^T rows == rows of given row-major matrix) fragment:
// rows row0..row0+15, K = k0..k0+31, from row-major [row][k], leading dim ld.
// Per ISA: lane<16 holds K {0..7,16..23}, lane>=16 holds K {8..15,24..31}.
static __device__ __forceinline__ v16bf frag_rows(const __bf16* p, int row0, int ld, int k0) {
  int l = laneid();
  const __bf16* q = p + (size_t)(row0 + (l & 15)) * ld + k0 + ((l >> 4) << 3);
  v16bf f;
#pragma unroll
  for (int j = 0; j < 16; ++j)
    f[j] = q[(j & 7) + ((j >> 3) << 4)];
  return f;
}

// B fragment where B^T rows are COLUMNS of a row-major [k][col] matrix.
static __device__ __forceinline__ v16bf frag_cols(const __bf16* p, int col0, int ld, int k0) {
  int l = laneid();
  int c  = col0 + (l & 15);
  int kb = k0 + ((l >> 4) << 3);
  v16bf f;
#pragma unroll
  for (int j = 0; j < 16; ++j)
    f[j] = p[(size_t)(kb + (j & 7) + ((j >> 3) << 4)) * ld + c];
  return f;
}

// ---------------- weight prep ----------------
__global__ void k_prep_weights(const float* thw, const float* phw, const float* gw,
                               const float* rw, const float* thb, const float* phb,
                               const float* gb, __bf16* wbf, __bf16* rbf, float* bias) {
  int t = threadIdx.x;
  for (int i = t; i < 64 * 128; i += blockDim.x) {
    wbf[i]             = (__bf16)thw[i];
    wbf[64 * 128 + i]  = (__bf16)phw[i];
    wbf[128 * 128 + i] = (__bf16)gw[i];
  }
  for (int i = t; i < 128 * 64; i += blockDim.x) rbf[i] = (__bf16)rw[i];
  for (int i = t; i < 64; i += blockDim.x) {
    bias[i] = thb[i]; bias[64 + i] = phb[i]; bias[128 + i] = gb[i];
  }
}

// ---------------- fused transpose + t/p/g projection ----------------
__global__ void __launch_bounds__(256)
k_proj(const float* __restrict__ h, const __bf16* __restrict__ wbf,
       const float* __restrict__ bias, __bf16* __restrict__ tpg) {
  __shared__ __bf16 lx[64][132];                    // 64 voxels x 128 channels
  __shared__ __align__(16) __bf16 lo[64][200];      // 64 voxels x 192 outputs
  const int n0 = blockIdx.x * 64;
  const int b  = n0 / SPAT;
  const int s0 = n0 - b * SPAT;
  const int t  = threadIdx.x;
  for (int i = t; i < 64 * 128; i += 256) {         // coalesced channels-first read
    int v = i & 63, c = i >> 6;
    lx[v][c] = (__bf16)h[(size_t)(b * IN_CHN + c) * SPAT + s0 + v];
  }
  if (blockIdx.x + 1 < gridDim.x) {                 // prefetch next X tile (one per 64B)
    for (int i = t; i < 128 * 4; i += 256) {
      int c = i >> 2, v = (i & 3) * 16;
      __builtin_prefetch(&h[(size_t)(b * IN_CHN + c) * SPAT + s0 + 64 + v], 0, 1);
    }
  }
  __syncthreads();
  const int wv = t >> 5;
  const int l  = laneid();
  for (int tile = wv; tile < 48; tile += 8) {       // 4 row-tiles x 12 out-tiles
    int m0 = (tile / 12) * 16;
    int o0 = (tile % 12) * 16;
    int col = o0 + (l & 15);
    float bv = bias[col];
    v8f acc;
#pragma unroll
    for (int r = 0; r < 8; ++r) acc[r] = bv;
#pragma unroll
    for (int k0 = 0; k0 < 128; k0 += 32) {
      v16bf a  = frag_rows(&lx[0][0], m0, 132, k0);
      v16bf bm = frag_rows(wbf, o0, 128, k0);       // rows of W == rows of B^T
      acc = wmma_bf16(a, bm, acc);
    }
    int mb = m0 + ((l >> 4) << 3);
#pragma unroll
    for (int r = 0; r < 8; ++r)
      lo[mb + r][col] = (__bf16)acc[r];
  }
  __syncthreads();
  for (int j = t; j < 64 * 24; j += 256) {          // contiguous b128 stores
    int r = j / 24, q = j - r * 24;
    *reinterpret_cast<uint4*>(tpg + (size_t)(n0 + r) * NPROJ + (q << 3)) =
        *reinterpret_cast<const uint4*>(&lo[r][q << 3]);
  }
}

// ---------------- per-line axis attention with online softmax ----------------
template <int DIR, bool FIRST>
__global__ void __launch_bounds__(128)
k_attn(const __bf16* __restrict__ tpg, float* __restrict__ Y,
       float* __restrict__ mbuf, float* __restrict__ lbuf) {
  __shared__ __align__(16) __bf16 lt[48][72];
  __shared__ __align__(16) __bf16 lp[48][72];
  __shared__ __align__(16) __bf16 lg[64][72];       // K-padded to 64 with zeros
  __shared__ float  lS[48][52];
  __shared__ __bf16 lE[48][72];
  __shared__ float  lscale[48];

  int idx = blockIdx.x;
  int b   = idx / (48 * 48);
  int rem = idx - b * 48 * 48;
  int p0 = rem / 48, p1 = rem % 48;
  int nbase, stride;
  if (DIR == 0)      { nbase = b * SPAT + p0 * 48 + p1;           stride = 48 * 48; }
  else if (DIR == 1) { nbase = b * SPAT + p0 * 48 * 48 + p1;      stride = 48; }
  else               { nbase = b * SPAT + p0 * 48 * 48 + p1 * 48; stride = 1; }

  const int t = threadIdx.x;
#if USE_ASYNC_LDS
  // Async DMA of t/p/g lines straight into LDS: 3 mats * 48 rows * 4 chunks(16B)
  for (int j = t; j < 3 * 48 * 4; j += 128) {
    int m  = j / (48 * 4);
    int rq = j - m * 48 * 4;
    int r  = rq >> 2, q = rq & 3;
    const __bf16* gsrc = tpg + (size_t)(nbase + r * stride) * NPROJ + m * 64 + (q << 4);
    __bf16* ldst = (m == 0) ? &lt[r][q << 4] : (m == 1) ? &lp[r][q << 4] : &lg[r][q << 4];
    __builtin_amdgcn_global_load_async_to_lds_b128((v4i_*)gsrc, (v4i_*)ldst, 0, 0);
  }
  for (int i = t + 48 * 64; i < 64 * 64; i += 128)  // zero K-pad rows of G
    lg[i >> 6][i & 63] = (__bf16)0.0f;
#if __has_builtin(__builtin_amdgcn_s_wait_asynccnt)
  __builtin_amdgcn_s_wait_asynccnt(0);
#else
  asm volatile("s_wait_asynccnt 0x0" ::: "memory");
#endif
#else
  for (int i = t; i < 48 * 64; i += 128) {
    int r = i >> 6, c = i & 63;
    const __bf16* src = tpg + (size_t)(nbase + r * stride) * NPROJ;
    lt[r][c] = src[c];
    lp[r][c] = src[64 + c];
    lg[r][c] = src[128 + c];
  }
  for (int i = t + 48 * 64; i < 64 * 64; i += 128)  // zero K-pad rows of G
    lg[i >> 6][i & 63] = (__bf16)0.0f;
#endif
  __syncthreads();

  const int wv = t >> 5;
  const int l  = laneid();
  // S = T @ P^T (48x48): 9 tiles
  for (int tile = wv; tile < 9; tile += 4) {
    int m0 = (tile / 3) * 16, c0 = (tile % 3) * 16;
    v8f acc = {};
#pragma unroll
    for (int k0 = 0; k0 < 64; k0 += 32) {
      v16bf a  = frag_rows(&lt[0][0], m0, 72, k0);
      v16bf bm = frag_rows(&lp[0][0], c0, 72, k0);  // B^T rows == rows of P
      acc = wmma_bf16(a, bm, acc);
    }
    int col = c0 + (l & 15);
    int mb  = m0 + ((l >> 4) << 3);
#pragma unroll
    for (int r = 0; r < 8; ++r) {
      float v = acc[r];
      if (DIR != 0 && (mb + r) == col) v = -3.0e38f; // diagonal mask (H/W)
      lS[mb + r][col] = v;
    }
  }
  __syncthreads();

  // online softmax chunk update (flash-style across the 3 directions)
  if (t < 48) {
    int n = nbase + t * stride;
    float mx = -3.4e38f;
    for (int j = 0; j < 48; ++j) mx = fmaxf(mx, lS[t][j]);
    float m_old, l_old;
    if (FIRST) { m_old = -3.4e38f; l_old = 0.0f; }
    else       { m_old = mbuf[n];  l_old = lbuf[n]; }
    float m_new = fmaxf(mx, m_old);
    float scale = FIRST ? 0.0f : __expf(m_old - m_new);
    float s = 0.0f;
    for (int j = 0; j < 48; ++j) {
      float e = __expf(lS[t][j] - m_new);
      s += e;
      lE[t][j] = (__bf16)e;
    }
    for (int j = 48; j < 64; ++j) lE[t][j] = (__bf16)0.0f;
    mbuf[n]   = m_new;
    lbuf[n]   = l_old * scale + s;
    lscale[t] = scale;
  }
  __syncthreads();

  // Ydir = E @ G (48x64): 12 tiles; accumulate into rescaled running Y
  for (int tile = wv; tile < 12; tile += 4) {
    int m0 = (tile >> 2) * 16, g0 = (tile & 3) * 16;
    v8f acc = {};
#pragma unroll
    for (int k0 = 0; k0 < 64; k0 += 32) {
      v16bf a  = frag_rows(&lE[0][0], m0, 72, k0);
      v16bf bm = frag_cols(&lg[0][0], g0, 72, k0);  // B^T rows == columns of G
      acc = wmma_bf16(a, bm, acc);
    }
    int col = g0 + (l & 15);
    int mb  = m0 + ((l >> 4) << 3);
#pragma unroll
    for (int r = 0; r < 8; ++r) {
      int row = mb + r;
      size_t o = (size_t)(nbase + row * stride) * 64 + col;
      if (FIRST) Y[o] = acc[r];
      else       Y[o] = Y[o] * lscale[row] + acc[r];
    }
  }
}

// ---------------- output projection + groupnorm partial stats ----------------
__global__ void __launch_bounds__(256)
k_cross(const float* __restrict__ Y, const float* __restrict__ lbuf,
        const __bf16* __restrict__ rbf, const float* __restrict__ rb,
        float* __restrict__ cross, float* __restrict__ gstats) {
  __shared__ __bf16 ly[64][72];
  __shared__ __align__(16) float lco[64][132];      // staged output tile
  __shared__ float  lsum[64];     // [0..31] sum, [32..63] sumsq per group
  const int n0 = blockIdx.x * 64;
  const int b  = n0 / SPAT;
  const int t  = threadIdx.x;
  if (t < 64) lsum[t] = 0.0f;
  for (int i = t; i < 64 * 64; i += 256) {
    int v = i >> 6, c = i & 63;
    float inv = 1.0f / lbuf[n0 + v];
    ly[v][c] = (__bf16)(Y[(size_t)(n0 + v) * 64 + c] * inv);
  }
  __syncthreads();
  const int wv = t >> 5, l = laneid();
  for (int tile = wv; tile < 32; tile += 8) {       // 4 row-tiles x 8 chan-tiles
    int m0 = (tile >> 3) * 16, c0 = (tile & 7) * 16;
    int col = c0 + (l & 15);
    float bv = rb[col];
    v8f acc;
#pragma unroll
    for (int r = 0; r < 8; ++r) acc[r] = bv;
#pragma unroll
    for (int k0 = 0; k0 < 64; k0 += 32) {
      v16bf a  = frag_rows(&ly[0][0], m0, 72, k0);
      v16bf bm = frag_rows(rbf, c0, 64, k0);        // rows of r_w == rows of B^T
      acc = wmma_bf16(a, bm, acc);
    }
    int mb = m0 + ((l >> 4) << 3);
    float s = 0.0f, sq = 0.0f;
#pragma unroll
    for (int r = 0; r < 8; ++r) {
      float v = acc[r];
      lco[mb + r][col] = v;
      s += v; sq += v * v;
    }
    int grp = col >> 2;
    atomicAdd(&lsum[grp], s);
    atomicAdd(&lsum[32 + grp], sq);
  }
  __syncthreads();
  for (int j = t; j < 64 * 32; j += 256) {          // contiguous b128 stores
    int r = j >> 5, q = j & 31;
    *reinterpret_cast<float4*>(cross + (size_t)(n0 + r) * IN_CHN + (q << 2)) =
        *reinterpret_cast<const float4*>(&lco[r][q << 2]);
  }
  if (t < 64) atomicAdd(&gstats[b * 64 + t], lsum[t]);
}

__global__ void k_zero(float* p, int n) {
  int i = blockIdx.x * blockDim.x + threadIdx.x;
  if (i < n) p[i] = 0.0f;
}

// ---------------- groupnorm apply + residual (channels-first out) ----------------
__global__ void __launch_bounds__(256)
k_gn_apply(const float* __restrict__ hin, const float* __restrict__ cross,
           const float* __restrict__ gstats, const float* __restrict__ gw,
           const float* __restrict__ gb, float* __restrict__ hout) {
  __shared__ float lc[64][129];
  __shared__ float lmu[32], lrs[32];
  const int n0 = blockIdx.x * 64;
  const int b  = n0 / SPAT;
  const int s0 = n0 - b * SPAT;
  const int t  = threadIdx.x;
  if (t < 32) {
    float cnt = 4.0f * (float)SPAT;
    float mu  = gstats[b * 64 + t] / cnt;
    float var = gstats[b * 64 + 32 + t] / cnt - mu * mu;
    lmu[t] = mu;
    lrs[t] = rsqrtf(var + EPSV);
  }
  for (int i = t; i < 64 * 128; i += 256) {         // coalesced channels-last read
    int v = i >> 7, c = i & 127;
    lc[v][c] = cross[(size_t)(n0 + v) * IN_CHN + c];
  }
  __syncthreads();
  for (int i = t; i < 64 * 128; i += 256) {         // coalesced channels-first write
    int v = i & 63, c = i >> 6;
    int grp = c >> 2;
    size_t o = (size_t)(b * IN_CHN + c) * SPAT + s0 + v;
    hout[o] = hin[o] + (lc[v][c] - lmu[grp]) * lrs[grp] * gw[c] + gb[c];
  }
}

// ---------------- batchnorm ----------------
__global__ void __launch_bounds__(256)
k_bn_reduce(const float* __restrict__ h, float* __restrict__ bnstats) {
  __shared__ float ssum[256], ssq[256];
  int c = blockIdx.x;
  size_t per = (size_t)NBATCH * SPAT / 8;
  float s = 0.0f, sq = 0.0f;
  for (size_t i = threadIdx.x; i < per; i += 256) {
    size_t e = (size_t)blockIdx.y * per + i;
    int bb = (int)(e / SPAT);
    size_t sidx = e - (size_t)bb * SPAT;
    float v = h[(size_t)(bb * IN_CHN + c) * SPAT + sidx];
    s += v; sq += v * v;
  }
  ssum[threadIdx.x] = s; ssq[threadIdx.x] = sq;
  __syncthreads();
  for (int o = 128; o > 0; o >>= 1) {
    if ((int)threadIdx.x < o) {
      ssum[threadIdx.x] += ssum[threadIdx.x + o];
      ssq[threadIdx.x]  += ssq[threadIdx.x + o];
    }
    __syncthreads();
  }
  if (threadIdx.x == 0) {
    atomicAdd(&bnstats[c * 2],     ssum[0]);
    atomicAdd(&bnstats[c * 2 + 1], ssq[0]);
  }
}

__global__ void k_bn_apply(const float* __restrict__ h, const float* __restrict__ bnstats,
                           const float* __restrict__ bw, const float* __restrict__ bb,
                           float* __restrict__ out) {
  size_t i = (size_t)blockIdx.x * blockDim.x + threadIdx.x;
  if (i >= (size_t)NVOX * IN_CHN) return;
  int c = (int)((i / SPAT) % IN_CHN);
  float cnt = (float)((size_t)NBATCH * SPAT);
  float mu  = bnstats[c * 2] / cnt;
  float var = bnstats[c * 2 + 1] / cnt - mu * mu;
  float v = (h[i] - mu) * rsqrtf(var + EPSV) * bw[c] + bb[c];
  out[i] = fmaxf(v, 0.0f);
}

extern "C" void kernel_launch(void* const* d_in, const int* in_sizes, int n_in,
                              void* d_out, int out_size, void* d_ws, size_t ws_size,
                              hipStream_t stream) {
  (void)in_sizes; (void)n_in; (void)out_size; (void)ws_size;
  const float* x   = (const float*)d_in[0];
  const float* thw = (const float*)d_in[1];
  const float* thb = (const float*)d_in[2];
  const float* phw = (const float*)d_in[3];
  const float* phb = (const float*)d_in[4];
  const float* gw_ = (const float*)d_in[5];
  const float* gb_ = (const float*)d_in[6];
  const float* rw  = (const float*)d_in[7];
  const float* rb  = (const float*)d_in[8];
  const float* gnw = (const float*)d_in[9];   // (2,128)
  const float* gnb = (const float*)d_in[10];
  const float* bnw = (const float*)d_in[11];
  const float* bnb = (const float*)d_in[12];
  float* out = (float*)d_out;

  char* ws = (char*)d_ws;
  size_t off = 0;
  auto alloc = [&](size_t bytes) -> char* {
    char* p = ws + off;
    off += (bytes + 255) & ~(size_t)255;
    return p;
  };
  __bf16* wbf    = (__bf16*)alloc((size_t)192 * 128 * 2);
  __bf16* rbf    = (__bf16*)alloc((size_t)128 * 64 * 2);
  float*  bias   = (float*) alloc(192 * 4);
  float*  gstats = (float*) alloc(128 * 4);
  float*  bnst   = (float*) alloc(256 * 4);
  __bf16* tpg    = (__bf16*)alloc((size_t)NVOX * 192 * 2);  // 85 MB
  float*  Ybuf   = (float*) alloc((size_t)NVOX * 64 * 4);   // 57 MB (adjacent)
  float*  mbuf   = (float*) alloc((size_t)NVOX * 4);
  float*  lbuf   = (float*) alloc((size_t)NVOX * 4);
  float*  cross  = (float*) alloc((size_t)NVOX * 128 * 4);  // 113 MB
  float*  h1     = (float*) alloc((size_t)NVOX * 128 * 4);  // 113 MB
  float*  h2     = (float*)tpg;  // reuse dead tpg+Y region (142 MB) for layer-2 output

  k_prep_weights<<<1, 256, 0, stream>>>(thw, phw, gw_, rw, thb, phb, gb_, wbf, rbf, bias);

  for (int layer = 0; layer < 2; ++layer) {
    const float* hin = (layer == 0) ? x : h1;
    float* hout      = (layer == 0) ? h1 : h2;
    k_proj<<<NVOX / 64, 256, 0, stream>>>(hin, wbf, bias, tpg);
    k_attn<0, true ><<<NBATCH * 48 * 48, 128, 0, stream>>>(tpg, Ybuf, mbuf, lbuf);
    k_attn<1, false><<<NBATCH * 48 * 48, 128, 0, stream>>>(tpg, Ybuf, mbuf, lbuf);
    k_attn<2, false><<<NBATCH * 48 * 48, 128, 0, stream>>>(tpg, Ybuf, mbuf, lbuf);
    k_zero<<<1, 128, 0, stream>>>(gstats, 128);
    k_cross<<<NVOX / 64, 256, 0, stream>>>(Ybuf, lbuf, rbf, rb, cross, gstats);
    k_gn_apply<<<NVOX / 64, 256, 0, stream>>>(hin, cross, gstats,
                                              gnw + layer * 128, gnb + layer * 128, hout);
  }
  k_zero<<<1, 256, 0, stream>>>(bnst, 256);
  k_bn_reduce<<<dim3(128, 8), 256, 0, stream>>>(h2, bnst);
  k_bn_apply<<<(NVOX * IN_CHN) / 256, 256, 0, stream>>>(h2, bnst, bnw, bnb, out);
}
